// OptimizedDSANLayer_16939351015987
// MI455X (gfx1250) — matrix operand order repaired and checked by reading, not compiled, not run on hardware
//
#include <hip/hip_runtime.h>

#define N_ATOMS 131072
#define NBLK    8192

typedef __attribute__((ext_vector_type(16))) _Float16 v16h;
typedef __attribute__((ext_vector_type(8)))  _Float16 v8h;
typedef __attribute__((ext_vector_type(4)))  _Float16 v4h;
typedef __attribute__((ext_vector_type(8)))  float    v8f;
typedef __attribute__((ext_vector_type(4)))  float    v4f;

static __device__ inline v8f wmma_f16(v16h a, v16h b, v8f c) {
  return __builtin_amdgcn_wmma_f32_16x16x32_f16(false, a, false, b, (short)0, c, false, false);
}

// A-fragment (16x32 f16) from LDS, row-major, stride in halfwords.
// Lane L: M = L&15, half = L>>4. halfwords 0..7 -> K = half*8+0..7,
// halfwords 8..15 -> K = 16+half*8+0..7 (per CDNA5 ISA 16-bit A layout).
static __device__ inline v16h load_a_frag(const _Float16* As, int strideA,
                                          int mtile, int kc, int lane) {
  int m = mtile * 16 + (lane & 15);
  int h = lane >> 4;
  const _Float16* row = As + m * strideA + kc * 32;
  v8h lo = *(const v8h*)(row + h * 8);
  v8h hi = *(const v8h*)(row + 16 + h * 8);
  v16h a;
#pragma unroll
  for (int i = 0; i < 8; ++i) { a[i] = lo[i]; a[i + 8] = hi[i]; }
  return a;
}

// B-fragment (32x16 f16) from pre-packed weights: lane n -> N = n&15,
// element i -> K = (n>>4)*8 + (i&7) + (i>>3)*16. Packed contiguously: 32B per lane.
static __device__ inline v16h load_b_frag(const _Float16* PW, int NT,
                                          int kc, int nt, int lane) {
  const _Float16* p = PW + (size_t)((kc * NT + nt) * 32 + lane) * 16;
  v8h lo = *(const v8h*)p;
  v8h hi = *(const v8h*)(p + 8);
  v16h b;
#pragma unroll
  for (int i = 0; i < 8; ++i) { b[i] = lo[i]; b[i + 8] = hi[i]; }
  return b;
}

static __device__ inline void zero8(v8f* a) {
#pragma unroll
  for (int j = 0; j < 8; ++j)
#pragma unroll
    for (int i = 0; i < 8; ++i) a[j][i] = 0.f;
}

static __device__ inline float waveSum(float v) {
#pragma unroll
  for (int off = 16; off > 0; off >>= 1) v += __shfl_xor(v, off, 32);
  return v;
}

// cooperative 64 x K f32 -> f16 LDS tile load (256 threads)
static __device__ inline void load_tile_f16(const float* __restrict__ src, int K,
                                            _Float16* As, int strideA,
                                            size_t rowBase, int tid) {
  int kv = K >> 2;
  int nv = 64 * kv;
  for (int i = tid; i < nv; i += 256) {
    int r = i / kv, c4 = i - r * kv;
    v4f v = *(const v4f*)(src + (rowBase + r) * (size_t)K + c4 * 4);
    v4h hh;
#pragma unroll
    for (int k = 0; k < 4; ++k) hh[k] = (_Float16)v[k];
    *(v4h*)(As + r * strideA + c4 * 4) = hh;
  }
}

// ---------------- prep kernels ----------------

__global__ void k_qvec(const float* __restrict__ seed, const float* __restrict__ wq,
                       const float* __restrict__ bq, float* __restrict__ q) {
  int n = threadIdx.x;
  float s = bq[n];
  for (int k = 0; k < 256; ++k) s += seed[k] * wq[k * 256 + n];
  q[n] = s;
}

// G = geom_w(16x64) @ cw[256:320,:] zero-padded to 32 rows; B = cb + geom_b @ cw[256:320,:]
__global__ void k_fusegeom(const float* __restrict__ gw, const float* __restrict__ gb,
                           const float* __restrict__ cw, const float* __restrict__ cb,
                           float* __restrict__ G, float* __restrict__ B) {
  int r = blockIdx.x, n = threadIdx.x;
  if (r < 16) {
    float s = 0.f;
    for (int j = 0; j < 64; ++j) s += gw[r * 64 + j] * cw[(size_t)(256 + j) * 256 + n];
    G[r * 256 + n] = s;
  } else {
    G[r * 256 + n] = 0.f;
  }
  if (r == 0) {
    float s = cb[n];
    for (int j = 0; j < 64; ++j) s += gb[j] * cw[(size_t)(256 + j) * 256 + n];
    B[n] = s;
  }
}

// pack f32 KxN (row-major, stride N) weight to f16 B-fragment layout
__global__ __launch_bounds__(256) void k_pack(const float* __restrict__ W,
                                              _Float16* __restrict__ P,
                                              int K, int Nout) {
  int fid = blockIdx.x * 8 + (threadIdx.x >> 5);
  int lane = threadIdx.x & 31;
  int NT = Nout >> 4;
  int kc = fid / NT, nt = fid - kc * NT;
  int n = nt * 16 + (lane & 15);
  int kb = kc * 32 + (lane >> 4) * 8;
  _Float16* d = P + ((size_t)fid * 32 + lane) * 16;
#pragma unroll
  for (int i = 0; i < 16; ++i) d[i] = (_Float16)W[(size_t)(kb + (i & 7) + (i >> 3) * 16) * Nout + n];
}

// ---------------- generic packed GEMM: out[M,256] = src[M,256] @ PW + bias ----------------

__global__ __launch_bounds__(256) void k_gemm256(const float* __restrict__ src,
                                                 const _Float16* __restrict__ PW,
                                                 const float* __restrict__ bias,
                                                 float* __restrict__ out, int relu) {
  __shared__ _Float16 As[64 * 264];
  int tid = threadIdx.x;
  size_t rowBase = (size_t)blockIdx.x * 64;
  load_tile_f16(src, 256, As, 264, rowBase, tid);
  __syncthreads();
  int wave = tid >> 5, lane = tid & 31;
  int mtile = wave & 3, ntBase = (wave >> 2) * 8;
  v8f acc[8]; zero8(acc);
  for (int kc = 0; kc < 8; ++kc) {
    v16h a = load_a_frag(As, 264, mtile, kc, lane);
#pragma unroll
    for (int j = 0; j < 8; ++j)
      acc[j] = wmma_f16(a, load_b_frag(PW, 16, kc, ntBase + j, lane), acc[j]);
  }
  int col = lane & 15, rh = (lane >> 4) * 8;
#pragma unroll
  for (int j = 0; j < 8; ++j) {
    int n = (ntBase + j) * 16 + col;
    float bb = bias[n];
#pragma unroll
    for (int v = 0; v < 8; ++v) {
      float x = acc[j][v] + bb;
      if (relu) x = fmaxf(x, 0.f);
      out[(rowBase + mtile * 16 + rh + v) * 256 + n] = x;
    }
  }
}

// ---------------- fused PMA: K/V GEMMs + segment softmax pool (4 blocks / WG) ----------------

__global__ __launch_bounds__(256) void k_pma_pool(const float* __restrict__ af,
                                                  const _Float16* __restrict__ PWK,
                                                  const _Float16* __restrict__ PWV,
                                                  const float* __restrict__ bk,
                                                  const float* __restrict__ bv,
                                                  const float* __restrict__ qv,
                                                  float* __restrict__ pooled) {
  __shared__ _Float16 As[64 * 264];
  __shared__ _Float16 Ks[64 * 264];
  __shared__ _Float16 Vs[64 * 264];
  __shared__ float Sc[8][64];
  int tid = threadIdx.x;
  size_t rowBase = (size_t)blockIdx.x * 64;
  load_tile_f16(af, 256, As, 264, rowBase, tid);
  __syncthreads();
  int wave = tid >> 5, lane = tid & 31;
  int mtile = wave & 3, ntBase = (wave >> 2) * 8;
  int col = lane & 15, rh = (lane >> 4) * 8;
  {
    v8f acc[8]; zero8(acc);
    for (int kc = 0; kc < 8; ++kc) {
      v16h a = load_a_frag(As, 264, mtile, kc, lane);
#pragma unroll
      for (int j = 0; j < 8; ++j)
        acc[j] = wmma_f16(a, load_b_frag(PWK, 16, kc, ntBase + j, lane), acc[j]);
    }
#pragma unroll
    for (int j = 0; j < 8; ++j) {
      int n = (ntBase + j) * 16 + col;
      float bb = bk[n];
#pragma unroll
      for (int v = 0; v < 8; ++v)
        Ks[(mtile * 16 + rh + v) * 264 + n] = (_Float16)(acc[j][v] + bb);
    }
  }
  {
    v8f acc[8]; zero8(acc);
    for (int kc = 0; kc < 8; ++kc) {
      v16h a = load_a_frag(As, 264, mtile, kc, lane);
#pragma unroll
      for (int j = 0; j < 8; ++j)
        acc[j] = wmma_f16(a, load_b_frag(PWV, 16, kc, ntBase + j, lane), acc[j]);
    }
#pragma unroll
    for (int j = 0; j < 8; ++j) {
      int n = (ntBase + j) * 16 + col;
      float bb = bv[n];
#pragma unroll
      for (int v = 0; v < 8; ++v)
        Vs[(mtile * 16 + rh + v) * 264 + n] = (_Float16)(acc[j][v] + bb);
    }
  }
  __syncthreads();
  for (int idx = tid; idx < 512; idx += 256) {
    int n = idx & 63, h = idx >> 6;
    float s = 0.f;
#pragma unroll
    for (int d = 0; d < 32; ++d) s += qv[h * 32 + d] * (float)Ks[n * 264 + h * 32 + d];
    Sc[h][n] = s * 0.17677669529663687f; // HD^-0.5
  }
  __syncthreads();
  int c = tid, hh = c >> 5;
  for (int bl = 0; bl < 4; ++bl) {
    float m = Sc[hh][bl * 16];
#pragma unroll
    for (int a2 = 1; a2 < 16; ++a2) m = fmaxf(m, Sc[hh][bl * 16 + a2]);
    float sum = 0.f, accv = 0.f;
#pragma unroll
    for (int a2 = 0; a2 < 16; ++a2) {
      float e = __expf(Sc[hh][bl * 16 + a2] - m);
      sum += e;
      accv += e * (float)Vs[(bl * 16 + a2) * 264 + c];
    }
    pooled[((size_t)blockIdx.x * 4 + bl) * 256 + c] = accv / sum;
  }
}

// ---------------- fused cross-attn: rbf + dual GEMMs + segment softmax ctx ----------------

__global__ __launch_bounds__(256) void k_cross(const float* __restrict__ af,
                                               const float* __restrict__ pos,
                                               const _Float16* __restrict__ PCK,
                                               const _Float16* __restrict__ PCV,
                                               const _Float16* __restrict__ PGK,
                                               const _Float16* __restrict__ PGV,
                                               const float* __restrict__ bk2,
                                               const float* __restrict__ bv2,
                                               const float* __restrict__ Qb,
                                               const float* __restrict__ centers,
                                               const float* __restrict__ widths,
                                               float* __restrict__ ctx) {
  __shared__ _Float16 As[64 * 264];
  __shared__ _Float16 Gs[64 * 40];
  __shared__ _Float16 Ks[64 * 264];
  __shared__ _Float16 Vs[64 * 264];
  __shared__ float red[256];
  __shared__ float Sc2[64];
  int tid = threadIdx.x;
  size_t rowBase = (size_t)blockIdx.x * 64;
  load_tile_f16(af, 256, As, 264, rowBase, tid);
  if (tid < 64) {
    int a = tid, bs = a & ~15;
    float cx = 0.f, cy = 0.f, cz = 0.f;
    for (int i = 0; i < 16; ++i) {
      const float* p = pos + (rowBase + bs + i) * 3;
      cx += p[0]; cy += p[1]; cz += p[2];
    }
    cx *= 0.0625f; cy *= 0.0625f; cz *= 0.0625f;
    const float* p = pos + (rowBase + a) * 3;
    float dx = p[0] - cx, dy = p[1] - cy, dz = p[2] - cz;
    float dist = sqrtf(dx * dx + dy * dy + dz * dz);
#pragma unroll
    for (int j = 0; j < 16; ++j) {
      float t = dist - centers[j];
      float w = widths[j];
      Gs[a * 40 + j] = (_Float16)__expf(-(t * t) / (2.f * w * w));
    }
#pragma unroll
    for (int j = 16; j < 32; ++j) Gs[a * 40 + j] = (_Float16)0.f;
  }
  __syncthreads();
  int wave = tid >> 5, lane = tid & 31;
  int mtile = wave & 3, ntBase = (wave >> 2) * 8;
  int col = lane & 15, rh = (lane >> 4) * 8;
  {
    v8f acc[8]; zero8(acc);
    for (int kc = 0; kc < 8; ++kc) {
      v16h a = load_a_frag(As, 264, mtile, kc, lane);
#pragma unroll
      for (int j = 0; j < 8; ++j)
        acc[j] = wmma_f16(a, load_b_frag(PCK, 16, kc, ntBase + j, lane), acc[j]);
    }
    v16h ag = load_a_frag(Gs, 40, mtile, 0, lane);
#pragma unroll
    for (int j = 0; j < 8; ++j)
      acc[j] = wmma_f16(ag, load_b_frag(PGK, 16, 0, ntBase + j, lane), acc[j]);
#pragma unroll
    for (int j = 0; j < 8; ++j) {
      int n = (ntBase + j) * 16 + col;
      float bb = bk2[n];
#pragma unroll
      for (int v = 0; v < 8; ++v)
        Ks[(mtile * 16 + rh + v) * 264 + n] = (_Float16)(acc[j][v] + bb);
    }
  }
  {
    v8f acc[8]; zero8(acc);
    for (int kc = 0; kc < 8; ++kc) {
      v16h a = load_a_frag(As, 264, mtile, kc, lane);
#pragma unroll
      for (int j = 0; j < 8; ++j)
        acc[j] = wmma_f16(a, load_b_frag(PCV, 16, kc, ntBase + j, lane), acc[j]);
    }
    v16h ag = load_a_frag(Gs, 40, mtile, 0, lane);
#pragma unroll
    for (int j = 0; j < 8; ++j)
      acc[j] = wmma_f16(ag, load_b_frag(PGV, 16, 0, ntBase + j, lane), acc[j]);
#pragma unroll
    for (int j = 0; j < 8; ++j) {
      int n = (ntBase + j) * 16 + col;
      float bb = bv2[n];
#pragma unroll
      for (int v = 0; v < 8; ++v)
        Vs[(mtile * 16 + rh + v) * 264 + n] = (_Float16)(acc[j][v] + bb);
    }
  }
  __syncthreads();
  {
    int a = tid >> 2, q4 = tid & 3;
    size_t blk = (size_t)blockIdx.x * 4 + (a >> 4);
    const float* Q = Qb + blk * 256;
    float s = 0.f;
    for (int c2 = q4 * 64; c2 < q4 * 64 + 64; ++c2) s += Q[c2] * (float)Ks[a * 264 + c2];
    red[a * 4 + q4] = s;
  }
  __syncthreads();
  if (tid < 64)
    Sc2[tid] = (red[tid * 4] + red[tid * 4 + 1] + red[tid * 4 + 2] + red[tid * 4 + 3]) * 0.0625f;
  __syncthreads();
  int c = tid;
  for (int bl = 0; bl < 4; ++bl) {
    float m = Sc2[bl * 16];
#pragma unroll
    for (int a2 = 1; a2 < 16; ++a2) m = fmaxf(m, Sc2[bl * 16 + a2]);
    float sum = 0.f, accv = 0.f;
#pragma unroll
    for (int a2 = 0; a2 < 16; ++a2) {
      float e = __expf(Sc2[bl * 16 + a2] - m);
      sum += e;
      accv += e * (float)Vs[(bl * 16 + a2) * 264 + c];
    }
    ctx[((size_t)blockIdx.x * 4 + bl) * 256 + c] = accv / sum;
  }
}

// ---------------- fused LN1 + FFN (residual written pre-LN2) ----------------

__global__ __launch_bounds__(256) void k_ffn(const float* __restrict__ af,
                                             const float* __restrict__ upd,
                                             const float* __restrict__ g1,
                                             const float* __restrict__ b1,
                                             const _Float16* __restrict__ PF1,
                                             const float* __restrict__ fb1,
                                             const _Float16* __restrict__ PF2,
                                             const float* __restrict__ fb2,
                                             float* __restrict__ out) {
  __shared__ _Float16 As[64 * 264];
  __shared__ _Float16 Hs[64 * 520];
  int tid = threadIdx.x, wave = tid >> 5, lane = tid & 31;
  size_t rowBase = (size_t)blockIdx.x * 64;
  // LN1(af + upd[bid]) -> f16 x1 tile
  for (int rr = 0; rr < 8; ++rr) {
    int r = wave * 8 + rr;
    size_t gr = rowBase + r;
    const float* arow = af + gr * 256;
    const float* urow = upd + (gr >> 4) * 256;
    int c0 = lane * 4, c1 = 128 + lane * 4;
    v4f x0 = *(const v4f*)(arow + c0) + *(const v4f*)(urow + c0);
    v4f x1 = *(const v4f*)(arow + c1) + *(const v4f*)(urow + c1);
    float s = x0[0] + x0[1] + x0[2] + x0[3] + x1[0] + x1[1] + x1[2] + x1[3];
    s = waveSum(s);
    float mu = s * (1.f / 256.f);
    float q = 0.f;
#pragma unroll
    for (int k = 0; k < 4; ++k) {
      float d0 = x0[k] - mu, d1 = x1[k] - mu;
      q += d0 * d0 + d1 * d1;
    }
    q = waveSum(q);
    float rs = rsqrtf(q * (1.f / 256.f) + 1e-5f);
    v4f gg0 = *(const v4f*)(g1 + c0), bb0 = *(const v4f*)(b1 + c0);
    v4f gg1 = *(const v4f*)(g1 + c1), bb1 = *(const v4f*)(b1 + c1);
    v4h y0, y1;
#pragma unroll
    for (int k = 0; k < 4; ++k) {
      y0[k] = (_Float16)((x0[k] - mu) * rs * gg0[k] + bb0[k]);
      y1[k] = (_Float16)((x1[k] - mu) * rs * gg1[k] + bb1[k]);
    }
    *(v4h*)(As + r * 264 + c0) = y0;
    *(v4h*)(As + r * 264 + c1) = y1;
  }
  __syncthreads();
  int mtile = wave & 3;
  int col = lane & 15, rh = (lane >> 4) * 8;
  // GEMM1: x1 @ ffn_w1 (+b1, relu) -> Hs f16 [64 x 512]
  for (int grp = 0; grp < 2; ++grp) {
    int ntBase = (wave >> 2) * 16 + grp * 8;
    v8f acc[8]; zero8(acc);
    for (int kc = 0; kc < 8; ++kc) {
      v16h a = load_a_frag(As, 264, mtile, kc, lane);
#pragma unroll
      for (int j = 0; j < 8; ++j)
        acc[j] = wmma_f16(a, load_b_frag(PF1, 32, kc, ntBase + j, lane), acc[j]);
    }
#pragma unroll
    for (int j = 0; j < 8; ++j) {
      int n = (ntBase + j) * 16 + col;
      float bb = fb1[n];
#pragma unroll
      for (int v = 0; v < 8; ++v)
        Hs[(mtile * 16 + rh + v) * 520 + n] = (_Float16)fmaxf(acc[j][v] + bb, 0.f);
    }
  }
  __syncthreads();
  // GEMM2: Hs @ ffn_w2 (+b2) + residual x1 -> out (pre-LN2)
  {
    int ntBase = (wave >> 2) * 8;
    v8f acc[8]; zero8(acc);
    for (int kc = 0; kc < 16; ++kc) {
      v16h a = load_a_frag(Hs, 520, mtile, kc, lane);
#pragma unroll
      for (int j = 0; j < 8; ++j)
        acc[j] = wmma_f16(a, load_b_frag(PF2, 16, kc, ntBase + j, lane), acc[j]);
    }
#pragma unroll
    for (int j = 0; j < 8; ++j) {
      int n = (ntBase + j) * 16 + col;
      float bb = fb2[n];
#pragma unroll
      for (int v = 0; v < 8; ++v) {
        int r = mtile * 16 + rh + v;
        out[(rowBase + r) * 256 + n] = (float)As[r * 264 + n] + acc[j][v] + bb;
      }
    }
  }
}

// ---------------- in-place LN2 (one wave per row) ----------------

__global__ __launch_bounds__(256) void k_ln2(float* __restrict__ x,
                                             const float* __restrict__ g,
                                             const float* __restrict__ b) {
  int wave = threadIdx.x >> 5, lane = threadIdx.x & 31;
  size_t row = (size_t)blockIdx.x * 8 + wave;
  float* p = x + row * 256;
  int c0 = lane * 4, c1 = 128 + lane * 4;
  v4f x0 = *(const v4f*)(p + c0);
  v4f x1 = *(const v4f*)(p + c1);
  float s = x0[0] + x0[1] + x0[2] + x0[3] + x1[0] + x1[1] + x1[2] + x1[3];
  s = waveSum(s);
  float mu = s * (1.f / 256.f);
  float q = 0.f;
#pragma unroll
  for (int k = 0; k < 4; ++k) {
    float d0 = x0[k] - mu, d1 = x1[k] - mu;
    q += d0 * d0 + d1 * d1;
  }
  q = waveSum(q);
  float rs = rsqrtf(q * (1.f / 256.f) + 1e-5f);
  v4f gg0 = *(const v4f*)(g + c0), bb0 = *(const v4f*)(b + c0);
  v4f gg1 = *(const v4f*)(g + c1), bb1 = *(const v4f*)(b + c1);
  v4f y0, y1;
#pragma unroll
  for (int k = 0; k < 4; ++k) {
    y0[k] = (x0[k] - mu) * rs * gg0[k] + bb0[k];
    y1[k] = (x1[k] - mu) * rs * gg1[k] + bb1[k];
  }
  *(v4f*)(p + c0) = y0;
  *(v4f*)(p + c1) = y1;
}

extern "C" void kernel_launch(void* const* d_in, const int* in_sizes, int n_in,
                              void* d_out, int out_size, void* d_ws, size_t ws_size,
                              hipStream_t stream) {
  (void)in_sizes; (void)n_in; (void)out_size; (void)ws_size;
  const float* af      = (const float*)d_in[0];
  const float* pos     = (const float*)d_in[1];
  const float* seed    = (const float*)d_in[4];
  const float* wq      = (const float*)d_in[5];
  const float* bq      = (const float*)d_in[6];
  const float* wk      = (const float*)d_in[7];
  const float* bk      = (const float*)d_in[8];
  const float* wv      = (const float*)d_in[9];
  const float* bv      = (const float*)d_in[10];
  const float* wo      = (const float*)d_in[11];
  const float* bo      = (const float*)d_in[12];
  const float* centers = (const float*)d_in[13];
  const float* widths  = (const float*)d_in[14];
  const float* geom_w  = (const float*)d_in[15];
  const float* geom_b  = (const float*)d_in[16];
  const float* cqw     = (const float*)d_in[17];
  const float* cqb     = (const float*)d_in[18];
  const float* ckw     = (const float*)d_in[19];
  const float* ckb     = (const float*)d_in[20];
  const float* cvw     = (const float*)d_in[21];
  const float* cvb     = (const float*)d_in[22];
  const float* m1w     = (const float*)d_in[23];
  const float* m1b     = (const float*)d_in[24];
  const float* m2w     = (const float*)d_in[25];
  const float* m2b     = (const float*)d_in[26];
  const float* f1w     = (const float*)d_in[27];
  const float* f1b     = (const float*)d_in[28];
  const float* f2w     = (const float*)d_in[29];
  const float* f2b     = (const float*)d_in[30];
  const float* g1      = (const float*)d_in[31];
  const float* b1      = (const float*)d_in[32];
  const float* g2      = (const float*)d_in[33];
  const float* b2      = (const float*)d_in[34];
  float* out = (float*)d_out;

  char* w = (char*)d_ws;
  size_t o = 0;
  auto alloc = [&](size_t bytes) -> char* {
    char* p = w + o;
    o += (bytes + 255) & ~(size_t)255;
    return p;
  };
  float*    QV  = (float*)alloc(256 * 4);
  float*    GKF = (float*)alloc(32 * 256 * 4);
  float*    GVF = (float*)alloc(32 * 256 * 4);
  float*    BK2 = (float*)alloc(256 * 4);
  float*    BV2 = (float*)alloc(256 * 4);
  _Float16* PWK = (_Float16*)alloc(256 * 256 * 2);
  _Float16* PWV = (_Float16*)alloc(256 * 256 * 2);
  _Float16* PWO = (_Float16*)alloc(256 * 256 * 2);
  _Float16* PCQ = (_Float16*)alloc(256 * 256 * 2);
  _Float16* PCK = (_Float16*)alloc(256 * 256 * 2);
  _Float16* PCV = (_Float16*)alloc(256 * 256 * 2);
  _Float16* PM1 = (_Float16*)alloc(256 * 256 * 2);
  _Float16* PM2 = (_Float16*)alloc(256 * 256 * 2);
  _Float16* PF1 = (_Float16*)alloc(256 * 512 * 2);
  _Float16* PF2 = (_Float16*)alloc(512 * 256 * 2);
  _Float16* PGK = (_Float16*)alloc(32 * 256 * 2);
  _Float16* PGV = (_Float16*)alloc(32 * 256 * 2);
  float* POOL = (float*)alloc((size_t)NBLK * 256 * 4);
  float* BF   = (float*)alloc((size_t)NBLK * 256 * 4);
  float* QB   = (float*)alloc((size_t)NBLK * 256 * 4);
  float* CT   = (float*)alloc((size_t)NBLK * 256 * 4);
  float* TMP  = (float*)alloc((size_t)NBLK * 256 * 4);
  float* UPD  = (float*)alloc((size_t)NBLK * 256 * 4);

  // prep
  k_qvec<<<1, 256, 0, stream>>>(seed, wq, bq, QV);
  k_fusegeom<<<32, 256, 0, stream>>>(geom_w, geom_b, ckw, ckb, GKF, BK2);
  k_fusegeom<<<32, 256, 0, stream>>>(geom_w, geom_b, cvw, cvb, GVF, BV2);
  k_pack<<<16, 256, 0, stream>>>(wk, PWK, 256, 256);
  k_pack<<<16, 256, 0, stream>>>(wv, PWV, 256, 256);
  k_pack<<<16, 256, 0, stream>>>(wo, PWO, 256, 256);
  k_pack<<<16, 256, 0, stream>>>(cqw, PCQ, 256, 256);
  k_pack<<<16, 256, 0, stream>>>(ckw, PCK, 256, 256);
  k_pack<<<16, 256, 0, stream>>>(cvw, PCV, 256, 256);
  k_pack<<<16, 256, 0, stream>>>(m1w, PM1, 256, 256);
  k_pack<<<16, 256, 0, stream>>>(m2w, PM2, 256, 256);
  k_pack<<<32, 256, 0, stream>>>(f1w, PF1, 256, 512);
  k_pack<<<32, 256, 0, stream>>>(f2w, PF2, 512, 256);
  k_pack<<<2, 256, 0, stream>>>(GKF, PGK, 32, 256);
  k_pack<<<2, 256, 0, stream>>>(GVF, PGV, 32, 256);

  // pipeline
  k_pma_pool<<<N_ATOMS / 64, 256, 0, stream>>>(af, PWK, PWV, bk, bv, QV, POOL);
  k_gemm256<<<NBLK / 64, 256, 0, stream>>>(POOL, PWO, bo, BF, 0);
  k_gemm256<<<NBLK / 64, 256, 0, stream>>>(BF, PCQ, cqb, QB, 0);
  k_cross<<<N_ATOMS / 64, 256, 0, stream>>>(af, pos, PCK, PCV, PGK, PGV, BK2, BV2,
                                            QB, centers, widths, CT);
  k_gemm256<<<NBLK / 64, 256, 0, stream>>>(CT, PM1, m1b, TMP, 1);
  k_gemm256<<<NBLK / 64, 256, 0, stream>>>(TMP, PM2, m2b, UPD, 0);
  k_ffn<<<N_ATOMS / 64, 256, 0, stream>>>(af, UPD, g1, b1, PF1, f1b, PF2, f2b, out);
  k_ln2<<<N_ATOMS / 8, 256, 0, stream>>>(out, g2, b2);
}
